// ExllamaLinear_46806553592209
// MI455X (gfx1250) — compile-verified, matrix operand order checked by reading, not run on hardware
//
#include <hip/hip_runtime.h>

typedef __attribute__((ext_vector_type(16))) _Float16 v16h;
typedef __attribute__((ext_vector_type(8)))  _Float16 v8h;
typedef __attribute__((ext_vector_type(4)))  _Float16 v4h;
typedef __attribute__((ext_vector_type(2)))  _Float16 h2;
typedef __attribute__((ext_vector_type(8)))  float    v8f;

#define IN_DIM  4096
#define OUT_DIM 11008
#define GROUP   128
#define MT      256          // block tile M (4 waves x 64 rows)
#define NT      128          // block tile N (2 waves x 64 cols)
#define KS      64           // K per LDS stage (2 WMMA K-chunks)
#define NCHUNK  (KS / 8)     // 8 K-chunks of 8 values per stage

// Dequant 2 nibbles (bits [3:0],[7:4] of t) -> fp16 pair (q - z) * s.
// 0x6400 | q == fp16(1024 + q) exactly; o2 = -(1024 + z) * s precomputed.
__device__ __forceinline__ h2 dq2(unsigned t, h2 s2, h2 o2) {
    unsigned p = (t & 0xFu) | ((t & 0xF0u) << 12) | 0x64006400u;
    h2 q = __builtin_bit_cast(h2, p);
    return q * s2 + o2;   // v_pk_fma_f16
}

__global__ __launch_bounds__(256)
void exllama_q4_wmma_kernel(const _Float16* __restrict__ x,
                            const int*      __restrict__ qweight,
                            const int*      __restrict__ qzeros,
                            const _Float16* __restrict__ scales,
                            const _Float16* __restrict__ bias,
                            _Float16*       __restrict__ out,
                            int M)
{
    // Dequantized B tile, double buffered: [stage][chunk][n][8 halves] = 32 KB
    __shared__ _Float16 sB[2 * NCHUNK * NT * 8];

    const int tid  = threadIdx.x;
    const int lane = tid & 31;
    const int wave = tid >> 5;
    const int wm   = wave & 3;     // 4 waves along M: 64 rows each
    const int wn   = wave >> 2;    // 2 waves along N: 64 cols each
    const int l    = lane & 15;
    const int lh   = lane >> 4;

    const int n0blk = blockIdx.x * NT;
    const int m0    = blockIdx.y * MT + wm * 64;

    // A per-lane row pointers, 4 M-fragments (lane-half lh owns K offset lh*8)
    const _Float16* arow[4];
#pragma unroll
    for (int f = 0; f < 4; ++f)
        arow[f] = x + (size_t)(m0 + f * 16 + l) * IN_DIM + lh * 8;

    // Producer mapping: wave 'pchunk' dequants one K-chunk of 8 for 4 columns/lane
    const int pchunk = tid >> 5;           // 0..7
    const int pn4    = (tid & 31) << 2;    // local col 0,4,...,124
    const int pn     = n0blk + pn4;        // global col

    v8f acc[4][4] = {};

    auto produce = [&](int kbase, int st) {
        const int kk = (kbase >> 3) + pchunk;       // qweight row
        const int g  = kbase / GROUP;               // quant group (stage-invariant)
        const uint4 w = *(const uint4*)(qweight + (size_t)kk * OUT_DIM + pn);
        const v4h  sv = *(const v4h*)(scales + (size_t)g * OUT_DIM + pn);
        unsigned   zz = ((unsigned)qzeros[(size_t)g * (OUT_DIM / 8) + (pn >> 3)])
                        >> ((pn & 4) * 4);
        if (kk + 2 * NCHUNK < IN_DIM / 8)   // prefetch 2 stages ahead
            __builtin_prefetch((const void*)(qweight +
                (size_t)(kk + 2 * NCHUNK) * OUT_DIM + pn), 0, 1);
        const unsigned wv[4] = { w.x, w.y, w.z, w.w };
        _Float16* dst = sB + ((size_t)(st * NCHUNK + pchunk) * NT + pn4) * 8;
#pragma unroll
        for (int i = 0; i < 4; ++i) {
            const _Float16 s = sv[i];
            const _Float16 o =
                (_Float16)(-(float)(1024 + ((zz >> (4 * i)) & 15))) * s;
            h2 s2 = { s, s }, o2 = { o, o };
            v8h r;
            ((h2*)&r)[0] = dq2(wv[i],       s2, o2);
            ((h2*)&r)[1] = dq2(wv[i] >> 8,  s2, o2);
            ((h2*)&r)[2] = dq2(wv[i] >> 16, s2, o2);
            ((h2*)&r)[3] = dq2(wv[i] >> 24, s2, o2);
            *(v8h*)(dst + i * 8) = r;       // ds_store_b128, contiguous
        }
    };

    produce(0, 0);
    __syncthreads();

    for (int kbase = 0; kbase < IN_DIM; kbase += KS) {
        const int st = (kbase >> 6) & 1;
        if (kbase + KS < IN_DIM)
            produce(kbase + KS, st ^ 1);    // fill other buffer while computing

#pragma unroll
        for (int c = 0; c < 2; ++c) {       // two WMMA K-chunks per stage
            const int kc = kbase + c * 32;

            v16h a[4];
#pragma unroll
            for (int f = 0; f < 4; ++f) {
                const v8h* p = (const v8h*)(arow[f] + kc);
                ((v8h*)&a[f])[0] = p[0];    // K kc..kc+7   (this lane-half)
                ((v8h*)&a[f])[1] = p[2];    // K kc+16..kc+23
            }

            // B fragments from LDS: lane-half lh reads chunks 4c+lh, 4c+lh+2
            const _Float16* sbase =
                sB + ((size_t)(st * NCHUNK + c * 4 + lh) * NT) * 8;
#pragma unroll
            for (int j = 0; j < 4; ++j) {
                const int nloc = wn * 64 + j * 16 + l;
                v16h b;
                ((v8h*)&b)[0] = *(const v8h*)(sbase + (size_t)nloc * 8);
                ((v8h*)&b)[1] = *(const v8h*)(sbase + (size_t)(2 * NT + nloc) * 8);

#pragma unroll
                for (int f = 0; f < 4; ++f) {
                    acc[f][j] = __builtin_amdgcn_wmma_f32_16x16x32_f16(
                        false, a[f], false, b, (short)0, acc[f][j], false, false);
                }
            }
        }
        __syncthreads();   // readers done with buf[st]; buf[st^1] stores visible
    }

    // Epilogue: fp32 acc -> fp16, then + bias (reference order).
    // Non-temporal stores: output is a 90MB write-once stream; keep the
    // x + qweight working set resident in the 192MB L2 instead.
#pragma unroll
    for (int f = 0; f < 4; ++f) {
        const int mrow = m0 + f * 16 + lh * 8;
#pragma unroll
        for (int j = 0; j < 4; ++j) {
            const int n = n0blk + wn * 64 + j * 16 + l;
            const _Float16 bv = bias[n];
#pragma unroll
            for (int r = 0; r < 8; ++r) {
                _Float16 v = (_Float16)acc[f][j][r] + bv;
                __builtin_nontemporal_store(
                    v, out + (size_t)(mrow + r) * OUT_DIM + n);
            }
        }
    }
}

extern "C" void kernel_launch(void* const* d_in, const int* in_sizes, int n_in,
                              void* d_out, int out_size, void* d_ws, size_t ws_size,
                              hipStream_t stream) {
    const _Float16* x       = (const _Float16*)d_in[0];
    const int*      qweight = (const int*)d_in[1];
    const int*      qzeros  = (const int*)d_in[2];
    const _Float16* scales  = (const _Float16*)d_in[3];
    const _Float16* bias    = (const _Float16*)d_in[4];
    _Float16*       out     = (_Float16*)d_out;

    const int M = in_sizes[0] / IN_DIM;          // 4096

    dim3 grid(OUT_DIM / NT, M / MT);             // 86 x 16 blocks
    exllama_q4_wmma_kernel<<<grid, 256, 0, stream>>>(x, qweight, qzeros,
                                                     scales, bias, out, M);
}
